// SmyrfSelfAttention_37512244363938
// MI455X (gfx1250) — matrix operand order repaired and checked by reading, not compiled, not run on hardware
//
#include <hip/hip_runtime.h>
#include <hip/hip_bf16.h>
#include <math.h>

// ---------------------------------------------------------------------------
// SMYRF self-attention for gfx1250 (CDNA5, wave32, WMMA bf16 16x16x32)
// ---------------------------------------------------------------------------

typedef __bf16 bf16_t;
typedef __attribute__((ext_vector_type(4)))  __bf16 v4bf;
typedef __attribute__((ext_vector_type(8)))  __bf16 v8bf;
typedef __attribute__((ext_vector_type(16))) __bf16 v16bf;
typedef __attribute__((ext_vector_type(8)))  float  v8f;

#define WMMA_BF16(a, b, c) \
  __builtin_amdgcn_wmma_f32_16x16x32_bf16(false, (a), false, (b), (short)0, (c), false, false)

constexpr int BATCH = 4;
constexpr int SEQ   = 4096;
constexpr int DM    = 1024;
constexpr int NH    = 16;
constexpr int DQ    = 64;
constexpr int CL    = 128;   // cluster size
constexpr int NHASH = 2;
constexpr int NCL   = 32;    // clusters
constexpr int ROT   = 16;
constexpr int BH    = BATCH * NH;      // 64
constexpr int MTOT  = BATCH * SEQ;     // 16384
constexpr float INV_SQRT_D = 0.125f;   // 1/sqrt(64)

// Build a v16bf A/B fragment from two contiguous 16-byte LDS halves.
__device__ inline v16bf frag16(const bf16_t* lo, const bf16_t* hi) {
  v8bf a = *reinterpret_cast<const v8bf*>(lo);
  v8bf b = *reinterpret_cast<const v8bf*>(hi);
  return __builtin_shufflevector(a, b, 0,1,2,3,4,5,6,7,8,9,10,11,12,13,14,15);
}

// Packed f32x4 -> bf16x4 (lowers to packed cvt + single ds_store_b64).
__device__ inline v4bf cvt4(float4 f) {
  v4bf t;
  t[0] = (bf16_t)f.x; t[1] = (bf16_t)f.y; t[2] = (bf16_t)f.z; t[3] = (bf16_t)f.w;
  return t;
}
__device__ inline v4bf cvt4s(float a, float b, float c, float d) {
  v4bf t;
  t[0] = (bf16_t)a; t[1] = (bf16_t)b; t[2] = (bf16_t)c; t[3] = (bf16_t)d;
  return t;
}

// ---------------------------------------------------------------------------
// Kernel 1: bias for output projection: biasO[m] = sum_n bo[n][m]
// ---------------------------------------------------------------------------
__global__ __launch_bounds__(256) void bias_sum_kernel(
    const float* __restrict__ bo, float* __restrict__ biasO) {
  int m = blockIdx.x * 256 + threadIdx.x;   // 0..1023
  float s = 0.f;
  #pragma unroll
  for (int n = 0; n < NH; n++) s += bo[n * DM + m];
  biasO[m] = s;
}

// ---------------------------------------------------------------------------
// Shared GEMM body: C[128x64] bf16-WMMA tile.
//   A: [*][lda] f32, rows mBase..mBase+127
//   B: [*][ldb] f32, cols nOff..nOff+63 (stored transposed in LDS)
// Per-wave: wm = M half (64 rows), wn = N subtile (16 cols), 4 accumulators.
// ---------------------------------------------------------------------------
__device__ __forceinline__ void gemm_tile_128x64(
    const float* __restrict__ A, int lda, int mBase,
    const float* __restrict__ Bg, int ldb, int nOff,
    bf16_t* As, bf16_t* Bst, int K, v8f acc[4]) {
  const int tid  = threadIdx.x;
  const int lane = tid & 31;
  const int w    = tid >> 5;
  const int wm   = w & 1;
  const int wn   = w >> 1;
  const int l15  = lane & 15;
  const int kh   = lane >> 4;

  // A loader: thread covers row tid/2, half of the 32-wide k-slab.
  const int rowA = tid >> 1;
  const int kofA = (tid & 1) * 16;
  // B loader: thread owns column n = tid&63 and 8 consecutive k values.
  const int nB   = tid & 63;
  const int kgB  = (tid >> 6) * 8;

  for (int kb = 0; kb < K; kb += 32) {
    // ---- A tile 128x32: f32 -> bf16, packed 8B LDS stores ----
    const float4* ap = reinterpret_cast<const float4*>(
        A + (size_t)(mBase + rowA) * lda + kb + kofA);
    #pragma unroll
    for (int u = 0; u < 4; u++)
      *reinterpret_cast<v4bf*>(&As[rowA * 40 + kofA + u * 4]) = cvt4(ap[u]);

    // ---- B tile 32x64 transposed -> Bst[n][k], packed 8B LDS stores ----
    {
      const float* bcol = Bg + (size_t)(kb + kgB) * ldb + nOff + nB;
      float v0 = bcol[0 * ldb], v1 = bcol[1 * ldb], v2 = bcol[2 * ldb], v3 = bcol[3 * ldb];
      float v4 = bcol[4 * ldb], v5 = bcol[5 * ldb], v6 = bcol[6 * ldb], v7 = bcol[7 * ldb];
      *reinterpret_cast<v4bf*>(&Bst[nB * 40 + kgB + 0]) = cvt4s(v0, v1, v2, v3);
      *reinterpret_cast<v4bf*>(&Bst[nB * 40 + kgB + 4]) = cvt4s(v4, v5, v6, v7);
    }

    // ---- prefetch next K-slab into L2 (global_prefetch_b8) ----
    if (kb + 32 < K) {
      __builtin_prefetch(A + (size_t)(mBase + rowA) * lda + kb + 32 + kofA, 0, 1);
      __builtin_prefetch(Bg + (size_t)(kb + 32 + kgB) * ldb + nOff + nB, 0, 1);
    }
    __syncthreads();

    const int ncol = wn * 16 + l15;
    v16bf bfr = frag16(&Bst[ncol * 40 + kh * 8], &Bst[ncol * 40 + 16 + kh * 8]);
    #pragma unroll
    for (int t = 0; t < 4; t++) {
      const int row = wm * 64 + t * 16 + l15;
      v16bf af = frag16(&As[row * 40 + kh * 8], &As[row * 40 + 16 + kh * 8]);
      acc[t] = WMMA_BF16(af, bfr, acc[t]);
    }
    __syncthreads();
  }
}

// ---------------------------------------------------------------------------
// Kernel 2: QKV projection.
// grid = (MTOT/128, 48)   blockIdx.y: which = z>>4 (0=q,1=k,2=v), head = z&15
// out layout: [B][H][S][64]
// ---------------------------------------------------------------------------
__global__ __launch_bounds__(256) void proj_qkv_kernel(
    const float* __restrict__ A,
    const float* __restrict__ wq, const float* __restrict__ bq,
    const float* __restrict__ wk, const float* __restrict__ bk,
    const float* __restrict__ wv, const float* __restrict__ bv,
    float* __restrict__ qo, float* __restrict__ ko, float* __restrict__ vo) {
  const int tid   = threadIdx.x;
  const int z     = blockIdx.y;
  const int which = z >> 4;
  const int head  = z & 15;
  const int mBase = blockIdx.x * 128;

  const float* Bg   = (which == 0 ? wq : which == 1 ? wk : wv) + (size_t)head * DM * DQ;
  const float* bias = (which == 0 ? bq : which == 1 ? bk : bv) + head * DQ;
  float*       outp = (which == 0 ? qo : which == 1 ? ko : vo);

  __shared__ __align__(16) bf16_t As[128 * 40];
  __shared__ __align__(16) bf16_t Bst[64 * 40];

  const int lane = tid & 31;
  const int w = tid >> 5, wm = w & 1, wn = w >> 1;
  const int l15 = lane & 15, kh = lane >> 4;

  v8f acc[4];
  #pragma unroll
  for (int t = 0; t < 4; t++)
    #pragma unroll
    for (int r = 0; r < 8; r++) acc[t][r] = 0.f;

  gemm_tile_128x64(A, DM, mBase, Bg, DQ, 0, As, Bst, DM, acc);

  const int ncol  = wn * 16 + l15;
  const float bv2 = bias[ncol];
  #pragma unroll
  for (int t = 0; t < 4; t++) {
    #pragma unroll
    for (int r = 0; r < 8; r++) {
      int m = mBase + wm * 64 + t * 16 + r + 8 * kh;
      int b = m >> 12;
      int s = m & 4095;
      outp[(((size_t)(b * NH + head)) * SEQ + s) * DQ + ncol] = acc[t][r] + bv2;
    }
  }
}

// ---------------------------------------------------------------------------
// Kernel 3: LSH bucket + stable counting sort.
// grid = (BH, NHASH, 2)  z: 0 = queries, 1 = keys.  256 threads.
// sortBuf layout: [qk][hash][bh][SEQ] ints.
// ---------------------------------------------------------------------------
__global__ __launch_bounds__(256) void lsh_sort_kernel(
    const float* __restrict__ qbuf, const float* __restrict__ kbuf,
    const float* __restrict__ rot, int* __restrict__ sortBuf) {
  const int bh = blockIdx.x, hash = blockIdx.y, qk = blockIdx.z;
  const int tid = threadIdx.x;
  const float* src = (qk == 0 ? qbuf : kbuf) + (size_t)bh * SEQ * DQ;
  const float* rp  = rot + (size_t)hash * DQ * ROT;

  __shared__ float rotS[DQ * ROT];       // 4 KB
  __shared__ int   hist[32 * 256];       // flat [bucket][thread], 32 KB
  __shared__ int   chunk[256];

  for (int i = tid; i < DQ * ROT; i += 256) rotS[i] = rp[i];
  #pragma unroll
  for (int b = 0; b < 32; b++) hist[b * 256 + tid] = 0;
  __syncthreads();

  int myBucket[16];
  #pragma unroll 1
  for (int j = 0; j < 16; j++) {
    int s = tid * 16 + j;
    const float* row = src + (size_t)s * DQ;
    float p[ROT];
    #pragma unroll
    for (int r = 0; r < ROT; r++) p[r] = 0.f;
    for (int d = 0; d < DQ; d++) {
      float v = row[d];
      #pragma unroll
      for (int r = 0; r < ROT; r++) p[r] += v * rotS[d * ROT + r];
    }
    // argmax over [p, -p] -> bucket in [0,32); first max wins (jnp.argmax)
    float best = -1e30f; int bi = 0;
    #pragma unroll
    for (int r = 0; r < 32; r++) {
      float v = (r < 16) ? p[r] : -p[r - 16];
      if (v > best) { best = v; bi = r; }
    }
    myBucket[j] = bi;
    hist[bi * 256 + tid] += 1;           // private column, no race
  }
  __syncthreads();

  // exclusive scan over 8192 flat entries (bucket-major, thread-minor)
  {
    int run = 0;
    #pragma unroll 1
    for (int i = 0; i < 32; i++) { int v = hist[tid * 32 + i]; hist[tid * 32 + i] = run; run += v; }
    chunk[tid] = run;
  }
  __syncthreads();
  if (tid == 0) {
    int run = 0;
    for (int i = 0; i < 256; i++) { int v = chunk[i]; chunk[i] = run; run += v; }
  }
  __syncthreads();
  {
    int add = chunk[tid];
    #pragma unroll 1
    for (int i = 0; i < 32; i++) hist[tid * 32 + i] += add;
  }
  __syncthreads();

  // stable scatter (token order preserved inside each thread's chunk)
  int* outp = sortBuf + (((size_t)(qk * NHASH + hash)) * BH + bh) * SEQ;
  #pragma unroll 1
  for (int j = 0; j < 16; j++) {
    int s = tid * 16 + j;
    int b = myBucket[j];
    int pos = hist[b * 256 + tid];
    hist[b * 256 + tid] = pos + 1;
    outp[pos] = s;
  }
}

// ---------------------------------------------------------------------------
// Kernel 4: clustered attention, one (cluster, bh, hash) per block.
// ---------------------------------------------------------------------------
__global__ __launch_bounds__(256) void lsh_attn_kernel(
    const float* __restrict__ qbuf, const float* __restrict__ kbuf,
    const float* __restrict__ vbuf, const int* __restrict__ sortBuf,
    float* __restrict__ obuf, float* __restrict__ lsebuf) {
  const int c = blockIdx.x, bh = blockIdx.y, hash = blockIdx.z;
  const int tid = threadIdx.x;

  __shared__ __align__(16) char smem[55296];
  int*    tokQ = reinterpret_cast<int*>(smem);              // 128 ints
  int*    tokK = reinterpret_cast<int*>(smem + 512);        // 128 ints
  bf16_t* Vst  = reinterpret_cast<bf16_t*>(smem + 1024);    // [64][136]  v^T
  bf16_t* Qs   = reinterpret_cast<bf16_t*>(smem + 18432);   // [128][72]
  bf16_t* Ks   = reinterpret_cast<bf16_t*>(smem + 36864);   // [128][72]
  bf16_t* Ps   = reinterpret_cast<bf16_t*>(smem + 18432);   // [128][136] aliases Qs/Ks

  const int* sq = sortBuf + (((size_t)(0 * NHASH + hash)) * BH + bh) * SEQ + c * CL;
  const int* sk = sortBuf + (((size_t)(1 * NHASH + hash)) * BH + bh) * SEQ + c * CL;
  if (tid < 128) tokQ[tid] = sq[tid];
  else           tokK[tid - 128] = sk[tid - 128];
  __syncthreads();

  // ---- gather q,k,v rows (f32 -> bf16); k scaled by 1/sqrt(d) ----
  {
    const float* qg = qbuf + (size_t)bh * SEQ * DQ;
    const float* kg = kbuf + (size_t)bh * SEQ * DQ;
    const float* vg = vbuf + (size_t)bh * SEQ * DQ;
    #pragma unroll 1
    for (int e = 0; e < 8; e++) {
      int idx = tid * 8 + e;            // 0..2047
      int row = idx >> 4;
      int seg = (idx & 15) * 4;
      {
        float4 f = *reinterpret_cast<const float4*>(qg + (size_t)tokQ[row] * DQ + seg);
        *reinterpret_cast<v4bf*>(&Qs[row * 72 + seg]) = cvt4(f);
      }
      {
        float4 f = *reinterpret_cast<const float4*>(kg + (size_t)tokK[row] * DQ + seg);
        *reinterpret_cast<v4bf*>(&Ks[row * 72 + seg]) =
            cvt4s(f.x * INV_SQRT_D, f.y * INV_SQRT_D, f.z * INV_SQRT_D, f.w * INV_SQRT_D);
      }
      {
        float4 f = *reinterpret_cast<const float4*>(vg + (size_t)tokK[row] * DQ + seg);
        Vst[(seg + 0) * 136 + row] = (bf16_t)f.x;
        Vst[(seg + 1) * 136 + row] = (bf16_t)f.y;
        Vst[(seg + 2) * 136 + row] = (bf16_t)f.z;
        Vst[(seg + 3) * 136 + row] = (bf16_t)f.w;
      }
    }
  }
  __syncthreads();

  const int w    = tid >> 5;     // wave owns rows w*16..w*16+15
  const int lane = tid & 31;
  const int l15  = lane & 15;
  const int kh   = lane >> 4;

  // ---- GEMM1: dots[128x128] = Q (128x64) * K^T ----
  v8f acc1[8];
  #pragma unroll
  for (int j = 0; j < 8; j++)
    #pragma unroll
    for (int r = 0; r < 8; r++) acc1[j][r] = 0.f;

  #pragma unroll
  for (int kb = 0; kb < 64; kb += 32) {
    const int row = w * 16 + l15;
    v16bf af = frag16(&Qs[row * 72 + kb + kh * 8], &Qs[row * 72 + kb + 16 + kh * 8]);
    #pragma unroll
    for (int j = 0; j < 8; j++) {
      const int col = j * 16 + l15;
      v16bf bfr = frag16(&Ks[col * 72 + kb + kh * 8], &Ks[col * 72 + kb + 16 + kh * 8]);
      acc1[j] = WMMA_BF16(af, bfr, acc1[j]);
    }
  }

  // ---- register softmax per row (rows split lanes0-15 / lanes16-31) ----
  float lseArr[8];
  #pragma unroll
  for (int r = 0; r < 8; r++) {
    float m = -1e30f;
    #pragma unroll
    for (int j = 0; j < 8; j++) m = fmaxf(m, acc1[j][r]);
    #pragma unroll
    for (int msk = 1; msk < 16; msk <<= 1) m = fmaxf(m, __shfl_xor(m, msk, 32));
    float sum = 0.f;
    #pragma unroll
    for (int j = 0; j < 8; j++) sum += expf(acc1[j][r] - m);
    #pragma unroll
    for (int msk = 1; msk < 16; msk <<= 1) sum += __shfl_xor(sum, msk, 32);
    float lse = m + logf(sum);
    lseArr[r] = lse;
    #pragma unroll
    for (int j = 0; j < 8; j++) acc1[j][r] = expf(acc1[j][r] - lse);
  }

  __syncthreads();   // everyone done reading Qs/Ks before Ps overwrites them

  // ---- write P (bf16) + lse ----
  #pragma unroll
  for (int r = 0; r < 8; r++) {
    const int row = w * 16 + r + 8 * kh;
    #pragma unroll
    for (int j = 0; j < 8; j++) Ps[row * 136 + j * 16 + l15] = (bf16_t)acc1[j][r];
  }
  if (l15 == 0) {
    #pragma unroll
    for (int r = 0; r < 8; r++) {
      const int row = w * 16 + r + 8 * kh;
      lsebuf[((size_t)(hash * BH + bh)) * SEQ + tokQ[row]] = lseArr[r];
    }
  }
  __syncthreads();

  // ---- GEMM2: o[128x64] = P (128x128) * V ----
  v8f acc2[4];
  #pragma unroll
  for (int j = 0; j < 4; j++)
    #pragma unroll
    for (int r = 0; r < 8; r++) acc2[j][r] = 0.f;

  #pragma unroll
  for (int kb = 0; kb < 128; kb += 32) {
    const int row = w * 16 + l15;
    v16bf af = frag16(&Ps[row * 136 + kb + kh * 8], &Ps[row * 136 + kb + 16 + kh * 8]);
    #pragma unroll
    for (int j = 0; j < 4; j++) {
      const int n = j * 16 + l15;
      v16bf bfr = frag16(&Vst[n * 136 + kb + kh * 8], &Vst[n * 136 + kb + 16 + kh * 8]);
      acc2[j] = WMMA_BF16(af, bfr, acc2[j]);
    }
  }

  // ---- scatter o to original token positions ----
  float* og = obuf + ((size_t)(hash * BH + bh)) * SEQ * DQ;
  #pragma unroll
  for (int j = 0; j < 4; j++) {
    const int n = j * 16 + l15;
    #pragma unroll
    for (int r = 0; r < 8; r++) {
      const int row = w * 16 + r + 8 * kh;
      og[(size_t)tokQ[row] * DQ + n] = acc2[j][r];
    }
  }
}

// ---------------------------------------------------------------------------
// Kernel 5: combine the two hash rounds with logsumexp weights; write into
// [B][S][h*64+d] so the output projection is a plain GEMM.
// ---------------------------------------------------------------------------
__global__ __launch_bounds__(256) void combine_kernel(
    const float* __restrict__ obuf, const float* __restrict__ lsebuf,
    float* __restrict__ comb) {
  size_t idx = (size_t)blockIdx.x * 256 + threadIdx.x;   // BH*SEQ*64
  int d  = (int)(idx & 63);
  int s  = (int)((idx >> 6) & 4095);
  int bh = (int)(idx >> 18);
  float l0 = lsebuf[((size_t)(0 * BH + bh)) * SEQ + s];
  float l1 = lsebuf[((size_t)(1 * BH + bh)) * SEQ + s];
  float m  = fmaxf(l0, l1);
  float tot = m + logf(expf(l0 - m) + expf(l1 - m));
  float w0 = expf(l0 - tot), w1 = expf(l1 - tot);
  float o0 = obuf[(((size_t)(0 * BH + bh)) * SEQ + s) * DQ + d];
  float o1 = obuf[(((size_t)(1 * BH + bh)) * SEQ + s) * DQ + d];
  int b = bh >> 4, h = bh & 15;
  comb[((size_t)b * SEQ + s) * DM + h * DQ + d] = w0 * o0 + w1 * o1;
}

// ---------------------------------------------------------------------------
// Kernel 6: output projection: out[16384][1024] = comb[16384][1024] x woFlat
// woFlat: wo[n][v][m] viewed as [k=n*64+v][1024].  + biasO.
// grid = (MTOT/128, DM/64)
// ---------------------------------------------------------------------------
__global__ __launch_bounds__(256) void out_proj_kernel(
    const float* __restrict__ A, const float* __restrict__ Bg,
    const float* __restrict__ biasO, float* __restrict__ Out) {
  const int tid   = threadIdx.x;
  const int mBase = blockIdx.x * 128;
  const int nBase = blockIdx.y * 64;

  __shared__ __align__(16) bf16_t As[128 * 40];
  __shared__ __align__(16) bf16_t Bst[64 * 40];

  const int lane = tid & 31;
  const int w = tid >> 5, wm = w & 1, wn = w >> 1;
  const int l15 = lane & 15, kh = lane >> 4;

  v8f acc[4];
  #pragma unroll
  for (int t = 0; t < 4; t++)
    #pragma unroll
    for (int r = 0; r < 8; r++) acc[t][r] = 0.f;

  gemm_tile_128x64(A, DM, mBase, Bg, DM, nBase, As, Bst, DM, acc);

  const int ncol = wn * 16 + l15;
  const float bv2 = biasO[nBase + ncol];
  #pragma unroll
  for (int t = 0; t < 4; t++) {
    #pragma unroll
    for (int r = 0; r < 8; r++) {
      int m = mBase + wm * 64 + t * 16 + r + 8 * kh;
      Out[(size_t)m * DM + nBase + ncol] = acc[t][r] + bv2;
    }
  }
}

// ---------------------------------------------------------------------------
// Launch
// ---------------------------------------------------------------------------
extern "C" void kernel_launch(void* const* d_in, const int* in_sizes, int n_in,
                              void* d_out, int out_size, void* d_ws, size_t ws_size,
                              hipStream_t stream) {
  (void)in_sizes; (void)n_in; (void)out_size; (void)ws_size;

  const float* hidden = (const float*)d_in[0];
  const float* wq  = (const float*)d_in[1];
  const float* bq  = (const float*)d_in[2];
  const float* wk  = (const float*)d_in[3];
  const float* bk  = (const float*)d_in[4];
  const float* wv  = (const float*)d_in[5];
  const float* bv  = (const float*)d_in[6];
  const float* wo  = (const float*)d_in[7];
  const float* bo  = (const float*)d_in[8];
  const float* rot = (const float*)d_in[9];

  // workspace layout (floats)
  float* ws    = (float*)d_ws;
  float* qb    = ws;                                   // [BH][S][64]
  float* kb2   = qb  + (size_t)BH * SEQ * DQ;
  float* vb    = kb2 + (size_t)BH * SEQ * DQ;
  float* ob    = vb  + (size_t)BH * SEQ * DQ;          // [2][BH][S][64]
  float* comb  = ob  + (size_t)NHASH * BH * SEQ * DQ;  // [16384][1024]
  float* lse   = comb + (size_t)MTOT * DM;             // [2][BH][S]
  float* biasO = lse + (size_t)NHASH * BH * SEQ;       // [1024]
  int*   sortB = (int*)(biasO + DM);                   // [2][2][BH][S]

  bias_sum_kernel<<<dim3(DM / 256), 256, 0, stream>>>(bo, biasO);

  proj_qkv_kernel<<<dim3(MTOT / 128, 3 * NH), 256, 0, stream>>>(
      hidden, wq, bq, wk, bk, wv, bv, qb, kb2, vb);

  lsh_sort_kernel<<<dim3(BH, NHASH, 2), 256, 0, stream>>>(qb, kb2, rot, sortB);

  lsh_attn_kernel<<<dim3(NCL, BH, NHASH), 256, 0, stream>>>(
      qb, kb2, vb, sortB, ob, lse);

  combine_kernel<<<dim3((unsigned)((size_t)BH * SEQ * DQ / 256)), 256, 0, stream>>>(
      ob, lse, comb);

  out_proj_kernel<<<dim3(MTOT / 128, DM / 64), 256, 0, stream>>>(
      comb, wo, biasO, (float*)d_out);
}
